// GraphConstructor_57561151701007
// MI455X (gfx1250) — matrix operand order repaired and checked by reading, not compiled, not run on hardware
//
#include <hip/hip_runtime.h>
#include <hip/hip_bf16.h>
#include <math.h>

typedef __attribute__((ext_vector_type(16))) _Float16 v16h;
typedef __attribute__((ext_vector_type(8)))  float    v8f;

#define N_NODES 8192
#define EMB     16
#define TOPK    16
#define NTILES  (N_NODES / 16)   // 512
#define NEG_BIG (-3.0e38f)

// Branchless fast tanh(3x): y = exp(-6|x|) in (0,1], t = (1-y)/(1+y) via
// hardware v_rcp_f32. Valid magnitude for any x; caller selects sign/zero.
__device__ __forceinline__ float tanh3_mag(float x) {
  float y = __expf(-6.0f * __builtin_fabsf(x));
  return (1.0f - y) * __builtin_amdgcn_rcpf(1.0f + y);
}

// ---------------------------------------------------------------------------
// Kernel 1: squash embeddings through tanh(3x), build packed vectors
//   P[n] = [m1(n) | m2(n)]   (A-side rows,   K = 32)
//   Q[n] = [m2(n) | -m1(n)]  (B-side columns, K = 32)
// so that  P_i . Q_j  =  m1_i.m2_j - m2_i.m1_j  = (S - S^T)[i,j].
// Also pre-swizzles f16 fragments into the CDNA5 WMMA per-lane layout:
//   A (16x32 f16): lane l -> row M = l&15, khalf = l>>4;
//       half h (VGPR v=h>>1, j=h&1): K = (v>=4?16:0) + 8*khalf + 2*(v&3) + j
//   B (32x16 f16): lane l -> column N = l&15, khalf = l>>4; half h -> same K as A
// ---------------------------------------------------------------------------
__global__ __launch_bounds__(256)
void prep_kernel(const float* __restrict__ e1, const float* __restrict__ e2,
                 _Float16* __restrict__ Ah, _Float16* __restrict__ Bh,
                 float* __restrict__ Pf, float* __restrict__ Qf) {
  int n = blockIdx.x * blockDim.x + threadIdx.x;
  if (n >= N_NODES) return;

  float P[32], Q[32];
#pragma unroll
  for (int k = 0; k < EMB; ++k) {
    float a = tanhf(3.0f * e1[n * EMB + k]);
    float b = tanhf(3.0f * e2[n * EMB + k]);
    P[k] = a;  P[16 + k] = b;
    Q[k] = b;  Q[16 + k] = -a;
  }
#pragma unroll
  for (int k = 0; k < 32; ++k) {
    Pf[n * 32 + k] = P[k];
    Qf[n * 32 + k] = Q[k];
  }

  const int ti = n >> 4;     // 16-row tile index
  const int r  = n & 15;     // row within tile

  // A fragment staging (this node feeds lanes r and r+16 of its tile)
#pragma unroll
  for (int khalf = 0; khalf < 2; ++khalf) {
    int lane = r + 16 * khalf;
#pragma unroll
    for (int h = 0; h < 16; ++h) {
      int v = h >> 1, j = h & 1;
      int k = ((v >= 4) ? 16 : 0) + 8 * khalf + ((v & 3) << 1) + j;
      Ah[(size_t)(ti * 32 + lane) * 16 + h] = (_Float16)P[k];
    }
  }
  // B fragment staging (this node is column r of tile ti; lanes r, r+16; same K order as A)
#pragma unroll
  for (int K = 0; K < 32; ++K) {
    Bh[(size_t)(ti * 32 + r + 16 * ((K >> 3) & 1)) * 16 + (K & 7) + ((K >> 4) << 3)] = (_Float16)Q[K];
  }
}

// ---------------------------------------------------------------------------
// Kernel 2: per-row top-k threshold on raw antisymmetric scores.
// One block per row: recompute the 8192 scores into LDS, then 16 rounds of
// block-wide argmax with invalidation; the 16th max is the threshold.
// ---------------------------------------------------------------------------
__global__ __launch_bounds__(256)
void thresh_kernel(const float* __restrict__ Pf, const float* __restrict__ Qf,
                   float* __restrict__ T) {
  __shared__ float sv[N_NODES];    // 32 KB: one full score row
  __shared__ float rmax[256];
  __shared__ int   ridx[256];
  __shared__ float prow[32];

  const int i = blockIdx.x;
  const int t = threadIdx.x;

  if (t < 32) prow[t] = Pf[i * 32 + t];
  __syncthreads();

  // Scores: each thread computes 32 of the 8192 entries (strided by 256)
  for (int m = 0; m < 32; ++m) {
    int j = t + (m << 8);
    const float* q = Qf + (size_t)j * 32;
    float acc = 0.0f;
#pragma unroll
    for (int k = 0; k < 32; ++k) acc = fmaf(prow[k], q[k], acc);
    sv[j] = acc;
  }
  __syncthreads();

  // 16 rounds of argmax-with-invalidation
  for (int r = 0; r < TOPK; ++r) {
    float lm = NEG_BIG;
    int   li = t;
    for (int m = 0; m < 32; ++m) {
      int j = t + (m << 8);
      float v = sv[j];
      if (v > lm) { lm = v; li = j; }
    }
    rmax[t] = lm;
    ridx[t] = li;
    __syncthreads();
    for (int s = 128; s > 0; s >>= 1) {
      if (t < s) {
        if (rmax[t + s] > rmax[t]) { rmax[t] = rmax[t + s]; ridx[t] = ridx[t + s]; }
      }
      __syncthreads();
    }
    if (t == 0) {
      if (r == TOPK - 1) T[i] = rmax[0];   // 16th largest raw score
      sv[ridx[0]] = NEG_BIG;               // invalidate winner
    }
    __syncthreads();
  }
}

// ---------------------------------------------------------------------------
// Kernel 3: WMMA GEMM + threshold mask + tanh, single (non-temporal) write
// of the 256 MB output. One block (8 waves) per 16-row strip; each wave
// sweeps 64 column tiles, one v_wmma_f32_16x16x32_f16 each (unrolled x2).
// D layout: lane l, VGPR v -> row = v + 8*(l>>4), col = l & 15.
// ---------------------------------------------------------------------------
__global__ __launch_bounds__(256)
void wmma_mask_kernel(const _Float16* __restrict__ Ah,
                      const _Float16* __restrict__ Bh,
                      const float* __restrict__ T,
                      float* __restrict__ out) {
  const int it    = blockIdx.x;          // i-tile (16 rows)
  const int lane  = threadIdx.x & 31;
  const int wave  = threadIdx.x >> 5;
  const int khalf = lane >> 4;
  const int i0    = it * 16;

  v16h a = *((const v16h*)Ah + (it * 32 + lane));

  float th[8];
#pragma unroll
  for (int v = 0; v < 8; ++v) th[v] = T[i0 + v + 8 * khalf];

  // Base output pointer for this lane's column within j-tile 'wave'
  // (row offset added per element; advances by 8*16 columns per iteration)
#pragma unroll 2
  for (int m = 0; m < NTILES / 8; ++m) {
    const int jt = wave + (m << 3);
    v16h b = *((const v16h*)Bh + (jt * 32 + lane));
    v8f  c = {};
    v8f  d = __builtin_amdgcn_wmma_f32_16x16x32_f16(
                 /*neg_a=*/false, a, /*neg_b=*/false, b,
                 /*c_mod=*/(short)0, c, /*reuse_a=*/false, /*reuse_b=*/false);

    const int col = jt * 16 + (lane & 15);
#pragma unroll
    for (int v = 0; v < 8; ++v) {
      float x    = d[v];
      float tval = tanh3_mag(x);                        // branchless
      bool  keep = (x > 0.0f) & (x >= th[v]);
      float val  = keep ? tval : 0.0f;                  // v_cndmask
      __builtin_nontemporal_store(
          val, out + (size_t)(i0 + v + 8 * khalf) * N_NODES + col);
    }
  }
}

// ---------------------------------------------------------------------------
extern "C" void kernel_launch(void* const* d_in, const int* in_sizes, int n_in,
                              void* d_out, int out_size, void* d_ws, size_t ws_size,
                              hipStream_t stream) {
  (void)in_sizes; (void)n_in; (void)out_size; (void)ws_size;

  const float* e1 = (const float*)d_in[0];
  const float* e2 = (const float*)d_in[1];
  float* out = (float*)d_out;

  char* ws = (char*)d_ws;
  _Float16* Ah = (_Float16*)(ws);                       // 512 KB: A fragments
  _Float16* Bh = (_Float16*)(ws + (size_t)(512 << 10)); // 512 KB: B fragments
  float*    Pf = (float*)(ws + (size_t)(1024 << 10));   // 1 MB: packed P (f32)
  float*    Qf = (float*)(ws + (size_t)(2048 << 10));   // 1 MB: packed Q (f32)
  float*    T  = (float*)(ws + (size_t)(3072 << 10));   // 32 KB: thresholds

  prep_kernel<<<N_NODES / 256, 256, 0, stream>>>(e1, e2, Ah, Bh, Pf, Qf);
  thresh_kernel<<<N_NODES, 256, 0, stream>>>(Pf, Qf, T);
  wmma_mask_kernel<<<NTILES, 256, 0, stream>>>(Ah, Bh, T, out);
}
